// EnKF_LSTM_58299886076036
// MI455X (gfx1250) — compile-verified, matrix operand order checked by reading, not run on hardware
//
#include <hip/hip_runtime.h>
#include <hip/hip_bf16.h>

typedef unsigned short u16;
typedef unsigned int   u32;
typedef __attribute__((ext_vector_type(16))) __bf16 v16bf;
typedef __attribute__((ext_vector_type(8)))  float  v8f;

#define LS   2
#define BS   64
#define TT   32
#define NIN  256
#define NH   256
#define NE   64
#define NENS 32
#define G4   1024

union FragAB { v16bf v; uint4 q[2]; };
union FragC  { v8f v; float f[8]; uint4 q[2]; };

__device__ __forceinline__ u16 f2bf(float f) {
    u32 u = __float_as_uint(f);
    u32 r = u + 0x7FFFu + ((u >> 16) & 1u);   // round-to-nearest-even
    return (u16)(r >> 16);
}
__device__ __forceinline__ float sigm(float x) { return 1.0f / (1.0f + __expf(-x)); }

__device__ __forceinline__ float nrand(u32 idx, u32 salt) {
    unsigned long long z = ((unsigned long long)idx << 32) | salt;
    z += 0x9E3779B97F4A7C15ull;
    z = (z ^ (z >> 30)) * 0xBF58476D1CE4E5B9ull;
    z = (z ^ (z >> 27)) * 0x94D049BB133111EBull;
    z ^= z >> 31;
    u32 u1 = (u32)z, u2 = (u32)(z >> 32);
    float f1 = ((float)u1 + 0.5f) * (1.0f / 4294967296.0f);
    float f2 = (float)u2 * (1.0f / 4294967296.0f);
    return sqrtf(fmaxf(-2.f * __logf(f1), 0.f)) * __cosf(6.28318530718f * f2);
}

// A-fragment (16x32 bf16, MxK) from bf16 LDS tile (row stride 256), ISA 16-bit A layout.
__device__ __forceinline__ void loadA_lds(FragAB& a, const u16* lds, int mt, int kt, int lane) {
    int row = mt * 16 + (lane & 15);
    int hb  = ((lane >> 4) & 1) * 8;
    const u16* p = lds + row * 256 + kt * 32 + hb;
    a.q[0] = *(const uint4*)(p);
    a.q[1] = *(const uint4*)(p + 16);
}
// B-fragment from fragment-linear packed weights: lane reads 32 contiguous bytes.
__device__ __forceinline__ void loadB_pack(FragAB& b, const u16* pk, int nT, int kt, int lane) {
    const u16* p = pk + (((nT * 8 + kt) * 32 + lane) << 4);
    b.q[0] = *(const uint4*)(p);
    b.q[1] = *(const uint4*)(p + 8);
}

// Async global->LDS stage of a 16 KB bf16 tile (8192 u16), 512 threads x 32 B each.
// INST_OFFSET applies to both LDS and global addresses (ISA 15.18.3), so the second
// half-tile reuses the same address registers with offset:8192.
__device__ __forceinline__ void async_stage16k(const u16* gbase, const u16* ldsDst, int tid) {
    u32 l0   = (u32)(size_t)ldsDst + (u32)tid * 16u;
    u32 voff = (u32)tid * 16u;
    asm volatile("global_load_async_to_lds_b128 %0, %1, %2"
                 :: "v"(l0), "v"(voff), "s"(gbase) : "memory");
    asm volatile("global_load_async_to_lds_b128 %0, %1, %2 offset:8192"
                 :: "v"(l0), "v"(voff), "s"(gbase) : "memory");
}
__device__ __forceinline__ void wait_async0() {
    asm volatile("s_wait_asynccnt 0x0" ::: "memory");
}

// ---------------- K0: pack Wi0/Wh0/Wi1/Wh1 into WMMA-B fragment order (bf16) ---------
__global__ __launch_bounds__(256) void k_pack(const float* __restrict__ Wih,
                                              const float* __restrict__ Whh,
                                              u16* __restrict__ packW) {
    int id = blockIdx.x * 256 + threadIdx.x;   // 4 * 262144 total
    int m    = id >> 18;                       // 0:Wi0 1:Wh0 2:Wi1 3:Wh1
    int li   = id & 262143;
    int e    = li & 15;
    int lane = (li >> 4) & 31;
    int kT   = (li >> 9) & 7;
    int nT   = li >> 12;
    int row  = nT * 16 + (lane & 15);
    int K    = kT * 32 + (e >> 3) * 16 + ((lane >> 4) & 1) * 8 + (e & 7);
    int layer = m >> 1;
    const float* W = (m & 1) ? Whh : Wih;
    packW[id] = f2bf(W[(layer * G4 + row) * NIN + K]);
}

// ---------------- K1: Gin0 = x @ Wi0^T + b0, stored in C-fragment order --------------
__global__ __launch_bounds__(512) void k_gates_x0(const float* __restrict__ x,
                                                  const float* __restrict__ bvec,
                                                  const u16* __restrict__ packWi0,
                                                  float* __restrict__ gin0) {
    int t = blockIdx.x >> 2, bt = blockIdx.x & 3;
    __shared__ __align__(16) u16 sA[16 * 256];
    int tid = threadIdx.x;
    for (int idx = tid; idx < 16 * 256; idx += 512) {
        int mm = idx >> 8, k = idx & 255;
        sA[idx] = f2bf(x[((bt * 16 + mm) * TT + t) * NIN + k]);
    }
    __syncthreads();
    int w = tid >> 5, lane = tid & 31;
    FragC acc[4];
#pragma unroll
    for (int g = 0; g < 4; ++g) {
        float bb = bvec[g * 256 + w * 16 + (lane & 15)];
#pragma unroll
        for (int r = 0; r < 8; ++r) acc[g].f[r] = bb;
    }
#pragma unroll
    for (int kt = 0; kt < 8; ++kt) {
        FragAB a; loadA_lds(a, sA, 0, kt, lane);
#pragma unroll
        for (int g = 0; g < 4; ++g) {
            FragAB b; loadB_pack(b, packWi0, g * 16 + w, kt, lane);
            acc[g].v = __builtin_amdgcn_wmma_f32_16x16x32_bf16(
                false, a.v, false, b.v, (short)0, acc[g].v, false, false);
        }
    }
#pragma unroll
    for (int g = 0; g < 4; ++g) {
        float* dst = gin0 + ((((t * 4 + bt) * 64 + (w * 4 + g)) * 32 + lane) << 3);
        *(uint4*)dst       = acc[g].q[0];
        *(uint4*)(dst + 4) = acc[g].q[1];
    }
}

// ---------------- K2: layer-0 recurrence, 2 M-tiles/wave (B frags reused 2x) ---------
__global__ __launch_bounds__(512) void k_lstm_l0(const float* __restrict__ enkf,
                                                 const u16* __restrict__ packWh0,
                                                 const float* __restrict__ gin0,
                                                 u16* __restrict__ hs0,
                                                 float* __restrict__ stateOut,
                                                 const float* qp, const float* ep) {
    int e = blockIdx.x >> 1, bt = blockIdx.x & 1;         // 32 batch rows per block
    __shared__ __align__(16) u16 sH[32 * 256];
    int tid = threadIdx.x;
    for (int idx = tid; idx < 8192; idx += 512) {
        int mm = idx >> 8, k = idx & 255;
        sH[idx] = f2bf(enkf[((bt * 32 + mm) * NH + k) * NE + e]);     // l=0, h half
    }
    int w = tid >> 5, lane = tid & 31, mh = (lane >> 4) & 1, nl = lane & 15;
    int n = w * 16 + nl;
    float cst[2][8], hcur[2][8];
#pragma unroll
    for (int mt = 0; mt < 2; ++mt)
#pragma unroll
        for (int r = 0; r < 8; ++r) {
            int ml = mt * 16 + r + mh * 8;
            cst[mt][r]  = enkf[((bt * 32 + ml) * NH + n) * NE + 32 + e];
            hcur[mt][r] = 0.f;
        }
    __syncthreads();
    for (int t = 0; t < TT; ++t) {
        FragC acc[2][4];
#pragma unroll
        for (int mt = 0; mt < 2; ++mt)
#pragma unroll
            for (int g = 0; g < 4; ++g) {
                const float* src = gin0 +
                    ((((t * 4 + bt * 2 + mt) * 64 + (w * 4 + g)) * 32 + lane) << 3);
                acc[mt][g].q[0] = *(const uint4*)src;
                acc[mt][g].q[1] = *(const uint4*)(src + 4);
            }
        if (t + 1 < TT)
            __builtin_prefetch(gin0 + (((((t + 1) * 4 + bt * 2) * 64 + (w * 4)) * 32 + lane) << 3), 0, 3);
#pragma unroll
        for (int kt = 0; kt < 8; ++kt) {
            FragAB a0, a1;
            loadA_lds(a0, sH, 0, kt, lane);
            loadA_lds(a1, sH, 1, kt, lane);
#pragma unroll
            for (int g = 0; g < 4; ++g) {
                FragAB b; loadB_pack(b, packWh0, g * 16 + w, kt, lane);
                acc[0][g].v = __builtin_amdgcn_wmma_f32_16x16x32_bf16(
                    false, a0.v, false, b.v, (short)0, acc[0][g].v, false, false);
                acc[1][g].v = __builtin_amdgcn_wmma_f32_16x16x32_bf16(
                    false, a1.v, false, b.v, (short)0, acc[1][g].v, false, false);
            }
        }
#pragma unroll
        for (int mt = 0; mt < 2; ++mt)
#pragma unroll
            for (int r = 0; r < 8; ++r) {
                float ig = sigm(acc[mt][0].f[r]);
                float fg = sigm(acc[mt][1].f[r]);
                float gg = tanhf(acc[mt][2].f[r]);
                float og = sigm(acc[mt][3].f[r]);
                cst[mt][r]  = fg * cst[mt][r] + ig * gg;
                hcur[mt][r] = og * tanhf(cst[mt][r]);
            }
        __syncthreads();                       // all waves done reading old h
#pragma unroll
        for (int mt = 0; mt < 2; ++mt)
#pragma unroll
            for (int r = 0; r < 8; ++r) {
                int ml = mt * 16 + r + mh * 8;
                sH[ml * 256 + n] = f2bf(hcur[mt][r]);
            }
        __syncthreads();                       // new h visible
        {   // coalesced bf16 copy of the 16KB h tile to Hs0 (layer-1 input)
            u16* dst = hs0 + ((e * TT + t) * BS + bt * 32) * NH;
            int idx = tid * 8;
            *(uint4*)(dst + idx)        = *(const uint4*)(sH + idx);
            *(uint4*)(dst + 4096 + idx) = *(const uint4*)(sH + 4096 + idx);
        }
    }
    float q = qp[0], eps = ep[0];
#pragma unroll
    for (int mt = 0; mt < 2; ++mt)
#pragma unroll
        for (int r = 0; r < 8; ++r) {
            int bg = bt * 32 + mt * 16 + r + mh * 8;
            u32 idx = ((0u * BS + bg) * NH + n) * NENS + e;
            stateOut[((0 * BS + bg) * NH + n) * NE + e]      = hcur[mt][r] + q   * nrand(idx, 0x11111111u);
            stateOut[((0 * BS + bg) * NH + n) * NE + 32 + e] = cst[mt][r]  + eps * nrand(idx, 0x22222222u);
        }
}

// ---------------- K3: layer-1 recurrence, fused input GEMM, async x staging ----------
__global__ __launch_bounds__(512) void k_lstm_l1(const float* __restrict__ enkf,
                                                 const u16* __restrict__ packWi1,
                                                 const u16* __restrict__ packWh1,
                                                 const float* __restrict__ bvec,
                                                 const u16* __restrict__ hs0,
                                                 float* __restrict__ stateOut,
                                                 const float* qp, const float* ep) {
    int e = blockIdx.x >> 1, bt = blockIdx.x & 1;
    __shared__ __align__(16) u16 sH[8192];
    __shared__ __align__(16) u16 sX[2][8192];   // double-buffered async staging
    int tid = threadIdx.x;
    for (int idx = tid; idx < 8192; idx += 512) {
        int mm = idx >> 8, k = idx & 255;
        sH[idx] = f2bf(enkf[((BS + bt * 32 + mm) * NH + k) * NE + e]);  // l=1, h half
    }
    int w = tid >> 5, lane = tid & 31, mh = (lane >> 4) & 1, nl = lane & 15;
    int n = w * 16 + nl;
    float cst[2][8], hcur[2][8], bvals[4];
#pragma unroll
    for (int mt = 0; mt < 2; ++mt)
#pragma unroll
        for (int r = 0; r < 8; ++r) {
            int ml = mt * 16 + r + mh * 8;
            cst[mt][r]  = enkf[((BS + bt * 32 + ml) * NH + n) * NE + 32 + e];
            hcur[mt][r] = 0.f;
        }
#pragma unroll
    for (int g = 0; g < 4; ++g) bvals[g] = bvec[G4 + g * 256 + w * 16 + nl];

    async_stage16k(hs0 + ((e * TT + 0) * BS + bt * 32) * NH, &sX[0][0], tid);  // t=0
    for (int t = 0; t < TT; ++t) {
        wait_async0();
        __syncthreads();                       // x[t&1] ready everywhere; prev h visible
        const u16* xb = &sX[t & 1][0];
        if (t + 1 < TT)
            async_stage16k(hs0 + ((e * TT + t + 1) * BS + bt * 32) * NH,
                           &sX[(t + 1) & 1][0], tid);
        FragC acc[2][4];
#pragma unroll
        for (int mt = 0; mt < 2; ++mt)
#pragma unroll
            for (int g = 0; g < 4; ++g)
#pragma unroll
                for (int r = 0; r < 8; ++r) acc[mt][g].f[r] = bvals[g];
#pragma unroll
        for (int kt = 0; kt < 8; ++kt) {       // x_t @ Wi1^T
            FragAB a0, a1;
            loadA_lds(a0, xb, 0, kt, lane);
            loadA_lds(a1, xb, 1, kt, lane);
#pragma unroll
            for (int g = 0; g < 4; ++g) {
                FragAB b; loadB_pack(b, packWi1, g * 16 + w, kt, lane);
                acc[0][g].v = __builtin_amdgcn_wmma_f32_16x16x32_bf16(
                    false, a0.v, false, b.v, (short)0, acc[0][g].v, false, false);
                acc[1][g].v = __builtin_amdgcn_wmma_f32_16x16x32_bf16(
                    false, a1.v, false, b.v, (short)0, acc[1][g].v, false, false);
            }
        }
#pragma unroll
        for (int kt = 0; kt < 8; ++kt) {       // h_prev @ Wh1^T
            FragAB a0, a1;
            loadA_lds(a0, sH, 0, kt, lane);
            loadA_lds(a1, sH, 1, kt, lane);
#pragma unroll
            for (int g = 0; g < 4; ++g) {
                FragAB b; loadB_pack(b, packWh1, g * 16 + w, kt, lane);
                acc[0][g].v = __builtin_amdgcn_wmma_f32_16x16x32_bf16(
                    false, a0.v, false, b.v, (short)0, acc[0][g].v, false, false);
                acc[1][g].v = __builtin_amdgcn_wmma_f32_16x16x32_bf16(
                    false, a1.v, false, b.v, (short)0, acc[1][g].v, false, false);
            }
        }
#pragma unroll
        for (int mt = 0; mt < 2; ++mt)
#pragma unroll
            for (int r = 0; r < 8; ++r) {
                float ig = sigm(acc[mt][0].f[r]);
                float fg = sigm(acc[mt][1].f[r]);
                float gg = tanhf(acc[mt][2].f[r]);
                float og = sigm(acc[mt][3].f[r]);
                cst[mt][r]  = fg * cst[mt][r] + ig * gg;
                hcur[mt][r] = og * tanhf(cst[mt][r]);
            }
        __syncthreads();                       // all reads of sH/sX done
#pragma unroll
        for (int mt = 0; mt < 2; ++mt)
#pragma unroll
            for (int r = 0; r < 8; ++r) {
                int ml = mt * 16 + r + mh * 8;
                sH[ml * 256 + n] = f2bf(hcur[mt][r]);
            }
    }
    float q = qp[0], eps = ep[0];
#pragma unroll
    for (int mt = 0; mt < 2; ++mt)
#pragma unroll
        for (int r = 0; r < 8; ++r) {
            int bg = bt * 32 + mt * 16 + r + mh * 8;
            u32 idx = ((1u * BS + bg) * NH + n) * NENS + e;
            stateOut[((BS + bg) * NH + n) * NE + e]      = hcur[mt][r] + q   * nrand(idx, 0x11111111u);
            stateOut[((BS + bg) * NH + n) * NE + 32 + e] = cst[mt][r]  + eps * nrand(idx, 0x22222222u);
        }
}

// ---------------- K4: EnKF small linear algebra (single block) -----------------------
__global__ __launch_bounds__(256) void k_enkf(const float* __restrict__ y,
                                              const float* __restrict__ Hvec,
                                              const float* rp,
                                              float* __restrict__ outBuf,
                                              float* __restrict__ Smat) {
    float* stateOut = outBuf + 64 + 4096;
    __shared__ float sHA[4096];        // hxi -> HA -> Y
    __shared__ float sW[8320];         // aug (64x130) -> sig-aug (64x66)
    __shared__ float sMean[64];
    __shared__ float sHvec[256];
    __shared__ float sPiv, sLog;
    int tid = threadIdx.x;
    float r = rp[0], r2 = r * r;
    for (int i = tid; i < 256; i += 256) sHvec[i] = Hvec[i];
    __syncthreads();
    for (int p = tid; p < 4096; p += 256) {    // hxi[b][ens] from last layer
        int b = p >> 6, ens = p & 63;
        const float* sp = stateOut + ((BS + b) * NH) * NE + ens;
        float s = 0.f;
        for (int nn = 0; nn < NH; ++nn) s += sp[nn * NE] * sHvec[nn];
        sHA[p] = s;
    }
    __syncthreads();
    if (tid < 64) { float m = 0; for (int j = 0; j < 64; ++j) m += sHA[tid * 64 + j]; sMean[tid] = m * (1.f / 64.f); }
    __syncthreads();
    for (int p = tid; p < 4096; p += 256) sHA[p] -= sMean[p >> 6];   // HA
    __syncthreads();
    for (int p = tid; p < 64 * 130; p += 256) {  // aug = [P | innov]
        int rr = p / 130, c = p % 130; float v = 0.f;
        if (c < 64) { for (int k = 0; k < 64; ++k) v += sHA[rr * 64 + k] * sHA[c * 64 + k];
                      v *= (1.f / 63.f); if (c == rr) v += r2; }
        else if (c < 128) { int j = c - 64; v = y[rr] - sMean[rr] - sHA[rr * 64 + j]; }
        sW[p] = v;
    }
    __syncthreads();
    for (int k = 0; k < 64; ++k) {               // Gauss-Jordan: P sol = innov
        if (tid == 0) sPiv = 1.f / sW[k * 130 + k];
        __syncthreads();
        if (tid < 130) sW[k * 130 + tid] *= sPiv;
        __syncthreads();
        if (tid < 64 && tid != k) {
            float f = sW[tid * 130 + k];
            for (int c = k; c < 128; ++c) sW[tid * 130 + c] -= f * sW[k * 130 + c];
        }
        __syncthreads();
    }
    for (int p = tid; p < 4096; p += 256) {      // S = HA^T sol  (for K5)
        int kk = p >> 6, N = p & 63; float s = 0.f;
        for (int c = 0; c < 64; ++c) s += sHA[c * 64 + kk] * sW[c * 130 + 64 + N];
        Smat[p] = s;
    }
    __syncthreads();
    for (int p = tid; p < 4096; p += 256) {      // Y = y - r^2 sol  (EnKF identity)
        int b = p >> 6, N = p & 63;
        sHA[p] = y[b] - r2 * sW[b * 130 + 64 + N];
    }
    __syncthreads();
    if (tid < 64) { float m = 0; for (int j = 0; j < 64; ++j) m += sHA[tid * 64 + j];
                    m *= (1.f / 64.f); sMean[tid] = m; outBuf[tid] = m; }   // filter_out
    __syncthreads();
    for (int p = tid; p < 4096; p += 256) sHA[p] -= sMean[p >> 6];          // Yc
    __syncthreads();
    for (int p = tid; p < 64 * 66; p += 256) {   // sig-aug (stride 66)
        int rr = p / 66, c = p % 66; float v = 0.f;
        if (c < 64) { for (int k = 0; k < 64; ++k) v += sHA[rr * 64 + k] * sHA[c * 64 + k];
                      v *= (1.f / 63.f); if (c == rr) v += r2;
                      outBuf[64 + rr * 64 + c] = v; }
        else if (c == 64) v = y[rr] - sMean[rr];
        sW[p] = v;
    }
    __syncthreads();
    if (tid == 0) sLog = 0.f;
    __syncthreads();
    for (int k = 0; k < 64; ++k) {               // GJ with logdet
        if (tid == 0) { float piv = sW[k * 66 + k]; sLog += __logf(fabsf(piv)); sPiv = 1.f / piv; }
        __syncthreads();
        if (tid < 66) sW[k * 66 + tid] *= sPiv;
        __syncthreads();
        if (tid < 64 && tid != k) {
            float f = sW[tid * 66 + k];
            for (int c = k; c <= 64; ++c) sW[tid * 66 + c] -= f * sW[k * 66 + c];
        }
        __syncthreads();
    }
    if (tid == 0) {
        float acc = 0.f;
        for (int rr = 0; rr < 64; ++rr) acc += (y[rr] - sMean[rr]) * sW[rr * 66 + 64];
        outBuf[64 + 4096 + LS * BS * NH * NE] = -0.5f * sLog - 0.5f * acc;   // ll
    }
}

// ---------------- K5: state += (state - mean) @ S / 63 ------------------------------
__global__ __launch_bounds__(256) void k_update(float* __restrict__ stateOut,
                                                const float* __restrict__ Smat) {
    __shared__ float sS[4096];
    __shared__ float sR[1024];
    __shared__ float sM[16];
    int tid = threadIdx.x;
    for (int p = tid; p < 4096; p += 256) sS[p] = Smat[p] * (1.f / 63.f);
    long rowBase = (long)blockIdx.x * 16;
    for (int p = tid; p < 1024; p += 256) sR[p] = stateOut[rowBase * 64 + p];
    __syncthreads();
    if (tid < 16) { float m = 0; for (int j = 0; j < 64; ++j) m += sR[tid * 64 + j]; sM[tid] = m * (1.f / 64.f); }
    __syncthreads();
    for (int p = tid; p < 1024; p += 256) {
        int rr = p >> 6, N = p & 63;
        float mu = sM[rr], u = 0.f;
        for (int k = 0; k < 64; ++k) u += (sR[rr * 64 + k] - mu) * sS[k * 64 + N];
        stateOut[rowBase * 64 + p] = sR[p] + u;
    }
}

extern "C" void kernel_launch(void* const* d_in, const int* in_sizes, int n_in,
                              void* d_out, int out_size, void* d_ws, size_t ws_size,
                              hipStream_t stream) {
    const float* x    = (const float*)d_in[0];
    const float* y    = (const float*)d_in[1];
    const float* enkf = (const float*)d_in[2];
    const float* Wih  = (const float*)d_in[3];
    const float* Whh  = (const float*)d_in[4];
    const float* bvec = (const float*)d_in[5];
    const float* Hv   = (const float*)d_in[6];
    const float* qp   = (const float*)d_in[7];
    const float* ep   = (const float*)d_in[8];
    const float* rp   = (const float*)d_in[9];
    float* out = (float*)d_out;
    char*  ws  = (char*)d_ws;

    u16*   packW = (u16*)ws;                                  // 2 MB (4 x 512KB)
    float* gin0  = (float*)(ws + (2u  << 20));                // 8 MB
    u16*   hs0   = (u16*)  (ws + (10u << 20));                // 32 MB
    float* Smat  = (float*)(ws + (42u << 20));                // 16 KB
    float* stateOut = out + 64 + 4096;

    k_pack    <<<4096,     256, 0, stream>>>(Wih, Whh, packW);
    k_gates_x0<<<TT * 4,   512, 0, stream>>>(x, bvec, packW, gin0);
    k_lstm_l0 <<<NENS * 2, 512, 0, stream>>>(enkf, packW + 1 * 262144, gin0, hs0, stateOut, qp, ep);
    k_lstm_l1 <<<NENS * 2, 512, 0, stream>>>(enkf, packW + 2 * 262144, packW + 3 * 262144,
                                             bvec, hs0, stateOut, qp, ep);
    k_enkf    <<<1,        256, 0, stream>>>(y, Hv, rp, out, Smat);
    k_update  <<<2048,     256, 0, stream>>>(stateOut, Smat);
}
